// VanillaRNN_89086211653812
// MI455X (gfx1250) — matrix operand order repaired and checked by reading, not compiled, not run on hardware
//
#include <hip/hip_runtime.h>

typedef __attribute__((ext_vector_type(16))) __bf16 v16bf;
typedef __attribute__((ext_vector_type(8)))  __bf16 v8bf;
typedef __attribute__((ext_vector_type(8)))  float  v8f;

#define RNN_V 256
#define RNN_H 1024
#define RNN_B 256
#define RNN_S 512
#define NT_H  (RNN_H / 16)   // 64 h column tiles
#define NT_Y  (RNN_V / 16)   // 16 y column tiles
#define KC    (RNN_H / 32)   // 32 K-chunks of 32
#define FRAG_ELEMS 512       // 32x16 bf16 per packed B fragment
#define WHH_P_ELEMS (NT_H * KC * FRAG_ELEMS)   // 1,048,576
#define WHY_P_ELEMS (NT_Y * KC * FRAG_ELEMS)   //   262,144

// ---------------------------------------------------------------------------
// Branch-free tanh. Uses the native CDNA5 V_TANH trans op when available
// (confirmed lowering to v_tanh_f32 on gfx1250); otherwise
// tanh(x) = sign(x) * (1 - e) / (1 + e), e = exp2(-2*log2(e)*|x|).
// All fallbacks are host-compilable (clang checks device bodies on host too).
// ---------------------------------------------------------------------------
__device__ __forceinline__ float fast_tanh(float x) {
#if defined(__HIP_DEVICE_COMPILE__) && __has_builtin(__builtin_amdgcn_tanhf)
  return __builtin_amdgcn_tanhf(x);
#else
  float ax = __builtin_fabsf(x);
  float e;
#if defined(__HIP_DEVICE_COMPILE__) && __has_builtin(__builtin_amdgcn_exp2f)
  e = __builtin_amdgcn_exp2f(ax * -2.885390081777927f);
#else
  e = __builtin_exp2f(ax * -2.885390081777927f);
#endif
  float r;
#if defined(__HIP_DEVICE_COMPILE__) && __has_builtin(__builtin_amdgcn_rcpf)
  r = __builtin_amdgcn_rcpf(1.0f + e);
#else
  r = 1.0f / (1.0f + e);
#endif
  float y = (1.0f - e) * r;
  return __builtin_copysignf(y, x);
#endif
}

// ---------------------------------------------------------------------------
// XOR-swizzled LDS layout for h (16 rows x 1024 bf16). 8-element (16B) chunks
// are XOR'd by (row & 7) so per-lane A-fragment b128 loads (16 distinct rows)
// land on distinct bank groups (2-way worst case = the b128 floor).
// ---------------------------------------------------------------------------
__device__ __forceinline__ int hswz(int row, int k) {
  return row * RNN_H + ((((k >> 3) ^ (row & 7)) << 3) | (k & 7));
}

// ---------------------------------------------------------------------------
// Pack Whh [1024x1024] and Why [1024x256] (fp32, row-major KxN) into bf16
// WMMA B-fragment order: frag f = nt*KC + kc holds the 32x16 tile
// B[kc*32 .. +31][nt*16 .. +15]; lane l owns 16 bf16 at l*16,
// element e -> K = kc*32 + 16*(l>>4) + e, N = nt*16 + (l&15).
// ---------------------------------------------------------------------------
__global__ __launch_bounds__(256) void pack_weights_kernel(
    const float* __restrict__ Whh, const float* __restrict__ Why,
    __bf16* __restrict__ WhhP, __bf16* __restrict__ WhyP) {
  int idx = blockIdx.x * 256 + threadIdx.x;
  if (idx >= WHH_P_ELEMS + WHY_P_ELEMS) return;
  const float* src;
  __bf16* dst;
  int N, lidx;
  if (idx < WHH_P_ELEMS) { src = Whh; dst = WhhP; N = RNN_H; lidx = idx; }
  else { src = Why; dst = WhyP; N = RNN_V; lidx = idx - WHH_P_ELEMS; }
  int e  = lidx & 15;
  int l  = (lidx >> 4) & 31;
  int f  = lidx >> 9;
  int kc = f & (KC - 1);
  int nt = f / KC;
  int k  = kc * 32 + ((l >> 4) * 16) + e;
  int n  = nt * 16 + (l & 15);
  dst[lidx] = (__bf16)src[(size_t)k * N + n];
}

// ---------------------------------------------------------------------------
// Persistent per-batch-block recurrence. grid = 16 blocks (16 batch rows
// each, rows are independent through h@Whh), 512 threads = 16 waves.
// h ping-pongs in LDS as bf16 (2 x 32KB).
// ---------------------------------------------------------------------------
__global__ __launch_bounds__(512) void rnn_seq_kernel(
    const int* __restrict__ x, const float* __restrict__ Wxh,
    const float* __restrict__ bh, const float* __restrict__ by,
    const __bf16* __restrict__ WhhP, const __bf16* __restrict__ WhyP,
    float* __restrict__ out, float* __restrict__ hfin) {
  __shared__ __bf16 hbuf[2][16 * RNN_H];

  const int b0    = blockIdx.x * 16;
  const int lane  = threadIdx.x & 31;
  const int wave  = threadIdx.x >> 5;   // 0..15
  const int nl    = lane & 15;          // N index within tile (also A row)
  const int khalf = lane >> 4;          // lane half -> K / M split

  // Decomposed swizzle: element offset of the 16B chunk holding
  // K = kc*32 + 16*khalf (+8) in row nl is (kc ^ sb2)*32 + lo_c (resp. hi_c)
  const int sb2  = (nl >> 2) & 1;
  const int lo_c = ((2 * khalf) ^ (nl & 3)) << 3;
  const int hi_c = ((2 * khalf + 1) ^ (nl & 3)) << 3;

  for (int i = threadIdx.x; i < 16 * RNN_H; i += 512)
    hbuf[0][i] = (__bf16)0.0f;          // h0 = 0
  __syncthreads();

  for (int t = 0; t < RNN_S; ++t) {
    const __bf16* __restrict__ hprev = hbuf[t & 1];
    __bf16* __restrict__ hnext       = hbuf[(t + 1) & 1];

    // token ids for the 8 batch rows this lane accumulates (C/D layout:
    // vgpr r, lane half khalf -> M = r + 8*khalf)
    int xi[8];
#pragma unroll
    for (int r = 0; r < 8; ++r)
      xi[r] = x[(size_t)(b0 + r + 8 * khalf) * RNN_S + t];

    // ---------------- h tiles: 4 per wave over NT_H = 64 ----------------
#pragma unroll 1
    for (int j = 0; j < 4; ++j) {
      const int nt = wave * 4 + j;
      const int n  = nt * 16 + nl;
      const float bhn = bh[n];
      v8f acc;
#pragma unroll
      for (int r = 0; r < 8; ++r)       // one-hot matmul == row gather
        acc[r] = Wxh[(size_t)xi[r] * RNN_H + n] + bhn;

      const __bf16* __restrict__ bp =
          WhhP + (size_t)nt * KC * FRAG_ELEMS + (size_t)lane * 16;
      const __bf16* __restrict__ alo_p = hprev + nl * RNN_H + lo_c;
      const __bf16* __restrict__ ahi_p = hprev + nl * RNN_H + hi_c;
#pragma unroll 4
      for (int kc = 0; kc < KC; ++kc) {
        const int koff = (kc ^ sb2) * 32;
        v8bf alo = *(const v8bf*)(alo_p + koff);
        v8bf ahi = *(const v8bf*)(ahi_p + koff);
        v16bf a  = __builtin_shufflevector(alo, ahi, 0, 1, 2, 3, 4, 5, 6, 7,
                                           8, 9, 10, 11, 12, 13, 14, 15);
        v16bf b  = *(const v16bf*)(bp + (size_t)kc * FRAG_ELEMS);
        acc = __builtin_amdgcn_wmma_f32_16x16x32_bf16(
            false, a, false, b, (short)0, acc, false, false);
      }
#pragma unroll
      for (int r = 0; r < 8; ++r)
        hnext[hswz(r + 8 * khalf, n)] = (__bf16)fast_tanh(acc[r]);
    }
    __syncthreads();   // hnext complete; old hprev is now dead

    // ---------------- y tile: 1 per wave over NT_Y = 16 -----------------
    {
      const int n = wave * 16 + nl;
      const float byn = by[n];
      v8f acc;
#pragma unroll
      for (int r = 0; r < 8; ++r) acc[r] = byn;

      const __bf16* __restrict__ bp =
          WhyP + (size_t)wave * KC * FRAG_ELEMS + (size_t)lane * 16;
      const __bf16* __restrict__ alo_p = hnext + nl * RNN_H + lo_c;
      const __bf16* __restrict__ ahi_p = hnext + nl * RNN_H + hi_c;
#pragma unroll 4
      for (int kc = 0; kc < KC; ++kc) {
        const int koff = (kc ^ sb2) * 32;
        v8bf alo = *(const v8bf*)(alo_p + koff);
        v8bf ahi = *(const v8bf*)(ahi_p + koff);
        v16bf a  = __builtin_shufflevector(alo, ahi, 0, 1, 2, 3, 4, 5, 6, 7,
                                           8, 9, 10, 11, 12, 13, 14, 15);
        v16bf b  = *(const v16bf*)(bp + (size_t)kc * FRAG_ELEMS);
        acc = __builtin_amdgcn_wmma_f32_16x16x32_bf16(
            false, a, false, b, (short)0, acc, false, false);
      }
#pragma unroll
      for (int r = 0; r < 8; ++r)
        out[((size_t)(b0 + r + 8 * khalf) * RNN_S + t) * RNN_V + n] = acc[r];
    }
    // No barrier needed here: next step writes only the now-dead buffer.
  }

  // ---- h_final: read last h (lives in hbuf[RNN_S & 1] == hbuf[0]) ----
  __syncthreads();
  const __bf16* __restrict__ hlast = hbuf[RNN_S & 1];
  for (int i = threadIdx.x; i < 16 * RNN_H; i += 512) {
    const int row = i >> 10;            // 0..15
    const int k   = i & (RNN_H - 1);    // 0..1023
    hfin[(size_t)(b0 + row) * RNN_H + k] = (float)hlast[hswz(row, k)];
  }
}

extern "C" void kernel_launch(void* const* d_in, const int* in_sizes, int n_in,
                              void* d_out, int out_size, void* d_ws,
                              size_t ws_size, hipStream_t stream) {
  const int*   x   = (const int*)d_in[0];
  const float* Wxh = (const float*)d_in[1];
  const float* Whh = (const float*)d_in[2];
  const float* Why = (const float*)d_in[3];
  const float* bh  = (const float*)d_in[4];
  const float* by  = (const float*)d_in[5];

  __bf16* WhhP = (__bf16*)d_ws;
  __bf16* WhyP = WhhP + WHH_P_ELEMS;

  float* out  = (float*)d_out;
  float* hfin = out + (size_t)RNN_B * RNN_S * RNN_V;

  const int pack_total  = WHH_P_ELEMS + WHY_P_ELEMS;
  const int pack_blocks = (pack_total + 255) / 256;
  pack_weights_kernel<<<pack_blocks, 256, 0, stream>>>(Whh, Why, WhhP, WhyP);

  rnn_seq_kernel<<<RNN_B / 16, 512, 0, stream>>>(x, Wxh, bh, by, WhhP, WhyP,
                                                 out, hfin);
}
